// Autoencoder_14697378087287
// MI455X (gfx1250) — matrix operand order repaired and checked by reading, not compiled, not run on hardware
//
#include <hip/hip_runtime.h>
#include <hip/hip_bf16.h>

// ---------------------------------------------------------------------------
// Shapes
// ---------------------------------------------------------------------------
#define S_LEN 512
#define BATCH 16
#define VOCAB 32003
#define VPAD  32256        // 63 * 512
#define EMB   512
#define HDIM  512
#define HHALF 256
#define SPAD  576          // 513 padded to 9*64 for ping-pong K loop

// ---------------------------------------------------------------------------
// WMMA fragment helpers (CDNA5 v_wmma_f32_16x16x32_f16, wave32)
//   A (16x32 f16): lane L -> row M = L&15; halves 0-15 = K[k0+kb .. +15],
//                  kb = (L>=16)?16:0
//   B (32x16 f16): lane L -> col N = L&15; halves 0-15 = K[k0+kb2 .. +15],
//                  kb2 = (L>=16)?16:0   (B stored as N rows x K cols, NT gemm)
//   D (16x16 f32): VGPR r, lane L -> M = r + 8*(L>=16), N = L&15
// ---------------------------------------------------------------------------
typedef _Float16 half8 __attribute__((ext_vector_type(8)));
typedef _Float16 v16h  __attribute__((ext_vector_type(16)));
typedef float    v8f   __attribute__((ext_vector_type(8)));

union V16U { v16h v; half8 h[2]; };

__device__ __forceinline__ v16h load_a16(const _Float16* ap, int k) {
  V16U u;
  u.h[0] = *(const half8*)(ap + k);
  u.h[1] = *(const half8*)(ap + k + 8);
  return u.v;
}

__device__ __forceinline__ v16h frag_a(const _Float16* A, int lda, int m, int k0,
                                       int hi, bool valid) {
  V16U u{};
  if (valid) {
    const _Float16* p = A + (size_t)m * lda + k0 + (hi ? 16 : 0);
    u.h[0] = *(const half8*)p;
    u.h[1] = *(const half8*)(p + 8);
  }
  return u.v;
}

__device__ __forceinline__ v16h frag_b(const _Float16* Bm, int ldb, int n, int k0,
                                       int hi) {
  return *(const v16h*)(Bm + (size_t)n * ldb + k0 + (hi ? 16 : 0));
}

__device__ __forceinline__ v8f wmma_f16(v16h a, v16h b, v8f c) {
  return __builtin_amdgcn_wmma_f32_16x16x32_f16(false, a, false, b, (short)0, c,
                                                false, false);
}

__device__ __forceinline__ float sigm(float x) { return 1.f / (1.f + __expf(-x)); }

// ---------------------------------------------------------------------------
// Generic NT GEMM: C[b][m][n] = sum_k A[b][m][k] * B[b][n][k] (+bias[n]) (ReLU?)
// A,B f16 K-contiguous rows; C f32 or f16. Grid: (ceil(M/16), N/64, batch),
// block = 32 (one wave). K multiple of 64, N multiple of 64.
// Structural ping-pong unroll-by-2: no fragment copies, no spills.
// A-row index clamped to M-1 for tail tiles (out-of-range D rows not stored).
// ---------------------------------------------------------------------------
__global__ void gemm_nt(const _Float16* A, long sA, int lda,
                        const _Float16* Bm, long sB, int ldb,
                        void* Cv, long sC, int ldc,
                        const float* bias,
                        int M, int K, int relu, int f16out) {
  const int lane = threadIdx.x;
  const int hi = lane >> 4, li = lane & 15;
  const int m0 = blockIdx.x * 16;
  const int n0 = blockIdx.y * 64;
  const _Float16* Ab = A + (size_t)blockIdx.z * sA;
  const _Float16* Bb = Bm + (size_t)blockIdx.z * sB;

  const int mrow = (m0 + li < M) ? (m0 + li) : (M - 1);
  const _Float16* ap  = Ab + (size_t)mrow * lda + (hi ? 16 : 0);
  const _Float16* bp0 = Bb + (size_t)(n0 + li) * ldb + (hi ? 16 : 0);
  const _Float16* bp1 = bp0 + (size_t)16 * ldb;
  const _Float16* bp2 = bp0 + (size_t)32 * ldb;
  const _Float16* bp3 = bp0 + (size_t)48 * ldb;

  v8f acc[4] = {};
  v16h a0 = load_a16(ap, 0);
  v16h b00 = *(const v16h*)(bp0), b01 = *(const v16h*)(bp1);
  v16h b02 = *(const v16h*)(bp2), b03 = *(const v16h*)(bp3);

  for (int k0 = 0; k0 < K; k0 += 64) {
    const int k1 = k0 + 32;
    v16h a1  = load_a16(ap, k1);
    v16h b10 = *(const v16h*)(bp0 + k1), b11 = *(const v16h*)(bp1 + k1);
    v16h b12 = *(const v16h*)(bp2 + k1), b13 = *(const v16h*)(bp3 + k1);
    acc[0] = wmma_f16(a0, b00, acc[0]);
    acc[1] = wmma_f16(a0, b01, acc[1]);
    acc[2] = wmma_f16(a0, b02, acc[2]);
    acc[3] = wmma_f16(a0, b03, acc[3]);
    const int k2 = k0 + 64;
    if (k2 < K) {  // uniform
      a0  = load_a16(ap, k2);
      b00 = *(const v16h*)(bp0 + k2); b01 = *(const v16h*)(bp1 + k2);
      b02 = *(const v16h*)(bp2 + k2); b03 = *(const v16h*)(bp3 + k2);
    }
    acc[0] = wmma_f16(a1, b10, acc[0]);
    acc[1] = wmma_f16(a1, b11, acc[1]);
    acc[2] = wmma_f16(a1, b12, acc[2]);
    acc[3] = wmma_f16(a1, b13, acc[3]);
  }

  const size_t cb = (size_t)blockIdx.z * sC;
#pragma unroll
  for (int j = 0; j < 4; ++j) {
    int n = n0 + 16 * j + li;
    float bv = bias ? bias[n] : 0.f;
#pragma unroll
    for (int r = 0; r < 8; ++r) {
      int m = m0 + hi * 8 + r;
      if (m < M) {
        float v = acc[j][r] + bv;
        if (relu) v = fmaxf(v, 0.f);
        if (f16out) ((_Float16*)Cv)[cb + (size_t)m * ldc + n] = (_Float16)v;
        else        ((float*)Cv)[cb + (size_t)m * ldc + n] = v;
      }
    }
  }
}

// ---------------------------------------------------------------------------
// Persistent LSTM: one workgroup, loops T serial steps. Per step:
//   g = xW[t] + h @ Whh^T  (WMMA from LDS-resident h, f16)
//   gate nonlinearities in f32, h/c kept in LDS.
// HSZ = hidden size (256 or 512). blockDim = HSZ -> HSZ/32 waves,
// N = 4*HSZ gate tiles = (HSZ/32)*8, i.e. 8 N-tiles per wave.
// ---------------------------------------------------------------------------
template <int HSZ>
__global__ void lstm_kernel(const float* xW, const _Float16* Whh,
                            int T, int t0, int tstep,
                            _Float16* outA, int ldA, int offA,
                            _Float16* outB, int ldB, int offB) {
  __shared__ _Float16 hs[16 * HSZ];
  __shared__ _Float16 gs[16 * 4 * HSZ];
  __shared__ float    cs[16 * HSZ];
  const int tid = threadIdx.x, lane = tid & 31, wave = tid >> 5;
  const int hi = lane >> 4, li = lane & 15;
  for (int i = tid; i < 16 * HSZ; i += HSZ) { hs[i] = (_Float16)0.f; cs[i] = 0.f; }
  __syncthreads();
  for (int s = 0; s < T; ++s) {
    const int t = t0 + s * tstep;
    v8f acc[8] = {};
    for (int k0 = 0; k0 < HSZ; k0 += 32) {
      v16h a = frag_a(hs, HSZ, li, k0, hi, true);
#pragma unroll
      for (int j = 0; j < 8; ++j) {
        v16h b = frag_b(Whh, HSZ, (wave * 8 + j) * 16 + li, k0, hi);
        acc[j] = wmma_f16(a, b, acc[j]);
      }
    }
#pragma unroll
    for (int j = 0; j < 8; ++j) {
      int n = (wave * 8 + j) * 16 + li;
#pragma unroll
      for (int r = 0; r < 8; ++r)
        gs[(hi * 8 + r) * 4 * HSZ + n] = (_Float16)acc[j][r];
    }
    __syncthreads();
    const float* xrow = xW + (size_t)t * 16 * 4 * HSZ;
    for (int i = tid; i < 16 * HSZ; i += HSZ) {
      int b = i / HSZ, jj = i - b * HSZ;
      const float* xg = xrow + (size_t)b * 4 * HSZ;
      const _Float16* gb = gs + (size_t)b * 4 * HSZ;
      float gi = (float)gb[jj]            + xg[jj];
      float gf = (float)gb[HSZ + jj]      + xg[HSZ + jj];
      float gc = (float)gb[2 * HSZ + jj]  + xg[2 * HSZ + jj];
      float go = (float)gb[3 * HSZ + jj]  + xg[3 * HSZ + jj];
      float c = sigm(gf) * cs[i] + sigm(gi) * tanhf(gc);
      float h = sigm(go) * tanhf(c);
      cs[i] = c;
      hs[i] = (_Float16)h;
      size_t orow = (size_t)t * 16 + b;
      outA[orow * ldA + offA + jj] = (_Float16)h;
      if (outB) outB[orow * ldB + offB + jj] = (_Float16)h;
    }
    __syncthreads();
  }
}

// ---------------------------------------------------------------------------
// Fused logits GEMM + online log-softmax + masked NLL accumulation.
// Block = 256 threads (8 waves), handles 16 token rows; loops V in 512-col
// chunks (each wave: 64 cols). hid tile staged in LDS via async-to-LDS copy.
// Ping-pong K loop: A from LDS, B from L2-resident Wo.
// ---------------------------------------------------------------------------
__global__ void logits_nll(const _Float16* hid, const _Float16* Wo,
                           const float* ob, const int* tgt, int Vpad,
                           float* acc2) {
  __shared__ _Float16 At[16 * 512];
  __shared__ float Ls[16 * 512];
  __shared__ float pm[16][17];
  __shared__ float ps[16][17];
  __shared__ float chunkM[16], run_m[16], run_s[16], tgtL[16];
  __shared__ int tgtN[16];
  const int tid = threadIdx.x, lane = tid & 31, wave = tid >> 5;
  const int hi = lane >> 4, li = lane & 15;
  const int row0 = blockIdx.x * 16;
  {
    // Async global->LDS staging of the 16x512 f16 hid tile (1024 x 16B).
    const unsigned int lds_base = (unsigned int)(unsigned long long)(&At[0]);
    const unsigned long long gbase =
        (unsigned long long)(const void*)(hid + (size_t)row0 * 512);
    for (int i = tid; i < 1024; i += 256) {
      unsigned int lds_addr = lds_base + (unsigned int)(i * 16);
      unsigned long long gaddr = gbase + (unsigned long long)i * 16ull;
      asm volatile("global_load_async_to_lds_b128 %0, %1, off"
                   :: "v"(lds_addr), "v"(gaddr) : "memory");
    }
    asm volatile("s_wait_asynccnt 0x0" ::: "memory");
  }
  if (tid < 16) {
    tgtN[tid] = tgt[row0 + tid];
    run_m[tid] = -1e30f; run_s[tid] = 0.f; tgtL[tid] = 0.f;
  }
  __syncthreads();
  const int row = tid >> 4, seg = tid & 15;
  const _Float16* ap = At + (size_t)li * 512 + (hi ? 16 : 0);  // LDS
  for (int c0 = 0; c0 < Vpad; c0 += 512) {
    const int n0 = c0 + wave * 64;
    const _Float16* bp0 = Wo + (size_t)(n0 + li) * 512 + (hi ? 16 : 0);
    const _Float16* bp1 = bp0 + (size_t)16 * 512;
    const _Float16* bp2 = bp0 + (size_t)32 * 512;
    const _Float16* bp3 = bp0 + (size_t)48 * 512;
    v8f acc[4] = {};
    v16h a0 = load_a16(ap, 0);
    v16h b00 = *(const v16h*)(bp0), b01 = *(const v16h*)(bp1);
    v16h b02 = *(const v16h*)(bp2), b03 = *(const v16h*)(bp3);
    for (int k0 = 0; k0 < 512; k0 += 64) {
      const int k1 = k0 + 32;
      v16h a1  = load_a16(ap, k1);
      v16h b10 = *(const v16h*)(bp0 + k1), b11 = *(const v16h*)(bp1 + k1);
      v16h b12 = *(const v16h*)(bp2 + k1), b13 = *(const v16h*)(bp3 + k1);
      acc[0] = wmma_f16(a0, b00, acc[0]);
      acc[1] = wmma_f16(a0, b01, acc[1]);
      acc[2] = wmma_f16(a0, b02, acc[2]);
      acc[3] = wmma_f16(a0, b03, acc[3]);
      const int k2 = k0 + 64;
      if (k2 < 512) {
        a0  = load_a16(ap, k2);
        b00 = *(const v16h*)(bp0 + k2); b01 = *(const v16h*)(bp1 + k2);
        b02 = *(const v16h*)(bp2 + k2); b03 = *(const v16h*)(bp3 + k2);
      }
      acc[0] = wmma_f16(a1, b10, acc[0]);
      acc[1] = wmma_f16(a1, b11, acc[1]);
      acc[2] = wmma_f16(a1, b12, acc[2]);
      acc[3] = wmma_f16(a1, b13, acc[3]);
    }
    __syncthreads();  // ensure previous chunk's Ls reads are done
#pragma unroll
    for (int j = 0; j < 4; ++j) {
      int ncol = wave * 64 + 16 * j + li;
      float bv = ob[n0 + 16 * j + li];
#pragma unroll
      for (int r = 0; r < 8; ++r)
        Ls[(hi * 8 + r) * 512 + ncol] = acc[j][r] + bv;
    }
    __syncthreads();
    float lm = -1e30f;
    for (int i = 0; i < 32; ++i) lm = fmaxf(lm, Ls[row * 512 + seg * 32 + i]);
    pm[row][seg] = lm;
    __syncthreads();
    if (tid < 16) {
      float cm = -1e30f;
      for (int i = 0; i < 16; ++i) cm = fmaxf(cm, pm[tid][i]);
      chunkM[tid] = cm;
      int tg = tgtN[tid];
      if (tg >= c0 && tg < c0 + 512) tgtL[tid] = Ls[tid * 512 + (tg - c0)];
    }
    __syncthreads();
    float cm = chunkM[row], sum = 0.f;
    for (int i = 0; i < 32; ++i) sum += __expf(Ls[row * 512 + seg * 32 + i] - cm);
    ps[row][seg] = sum;
    __syncthreads();
    if (tid < 16) {
      float csum = 0.f;
      for (int i = 0; i < 16; ++i) csum += ps[tid][i];
      float cmv = chunkM[tid];
      float mo = run_m[tid], mn = fmaxf(mo, cmv);
      run_s[tid] = run_s[tid] * __expf(mo - mn) + csum * __expf(cmv - mn);
      run_m[tid] = mn;
    }
  }
  __syncthreads();
  if (tid < 16) {
    int tg = tgtN[tid];
    if (tg != 0) {
      float nll = -(tgtL[tid] - run_m[tid] - __logf(run_s[tid]));
      atomicAdd(&acc2[0], nll);
      atomicAdd(&acc2[1], 1.0f);
    }
  }
}

// ---------------------------------------------------------------------------
// Small utility kernels
// ---------------------------------------------------------------------------
__global__ void f32_to_f16_k(const float* s, _Float16* d, size_t n) {
  size_t i = (size_t)blockIdx.x * blockDim.x + threadIdx.x;
  if (i < n) d[i] = (_Float16)s[i];
}
__global__ void zero_f16_k(_Float16* d, size_t n) {
  size_t i = (size_t)blockIdx.x * blockDim.x + threadIdx.x;
  if (i < n) d[i] = (_Float16)0.f;
}
__global__ void pad_bias_k(const float* s, float* d, int V, int Vp) {
  int i = blockIdx.x * blockDim.x + threadIdx.x;
  if (i < Vp) d[i] = (i < V) ? s[i] : -1e30f;
}
__global__ void zero2_k(float* a) { if (threadIdx.x < 2) a[threadIdx.x] = 0.f; }
__global__ void finalize_k(const float* a, float* o) { o[0] = a[0] / a[1]; }

// embed: row r = t*16+b; id = (t==0) ? 0 : ids[(t-1)*16+b]; f32 -> f16
__global__ void gather_emb_k(const float* emb, const int* ids, int T,
                             _Float16* out) {
  int r = blockIdx.x;
  if (r >= T * 16) return;
  int t = r >> 4, b = r & 15;
  int id = (t == 0) ? 0 : ids[(t - 1) * 16 + b];
  const float* src = emb + (size_t)id * 512;
  _Float16* d = out + (size_t)r * 512;
  for (int e = threadIdx.x; e < 512; e += blockDim.x) d[e] = (_Float16)src[e];
}

// attn softmax over s (513) for each (b,t); writes transposed padded f16 rows
__global__ void attn_softmax_k(const float* scores, _Float16* attnT) {
  int idx = blockIdx.x * blockDim.x + threadIdx.x;
  if (idx >= 16 * 512) return;
  int b = idx >> 9, t = idx & 511;
  const float* sc = scores + (size_t)b * 513 * 512 + t;
  float m = -1e30f;
  for (int s = 0; s < 513; ++s) m = fmaxf(m, sc[(size_t)s * 512]);
  float sum = 0.f;
  for (int s = 0; s < 513; ++s) sum += __expf(sc[(size_t)s * 512] - m);
  float inv = 1.f / sum;
  _Float16* o = attnT + (size_t)idx * SPAD;
  for (int s = 0; s < 513; ++s)
    o[s] = (_Float16)(__expf(sc[(size_t)s * 512] - m) * inv);
  for (int s = 513; s < SPAD; ++s) o[s] = (_Float16)0.f;
}

// encT[b][h][s(pad SPAD)] = enc16[(s*16+b)*512 + h]
__global__ void transpose_enc_k(const _Float16* enc, _Float16* encT) {
  size_t idx = (size_t)blockIdx.x * blockDim.x + threadIdx.x;
  const size_t total = (size_t)16 * 512 * SPAD;
  if (idx >= total) return;
  int s = (int)(idx % SPAD);
  int h = (int)((idx / SPAD) % 512);
  int b = (int)(idx / (SPAD * 512));
  encT[idx] = (s < 513) ? enc[((size_t)s * 16 + b) * 512 + h] : (_Float16)0.f;
}

// ---------------------------------------------------------------------------
// Host launch
// ---------------------------------------------------------------------------
extern "C" void kernel_launch(void* const* d_in, const int* in_sizes, int n_in,
                              void* d_out, int out_size, void* d_ws,
                              size_t ws_size, hipStream_t stream) {
  (void)in_sizes; (void)n_in; (void)out_size; (void)ws_size;
  const int*   numeric   = (const int*)d_in[0];
  const int*   noised    = (const int*)d_in[1];
  const float* emb       = (const float*)d_in[2];
  const float* enc_Wih_f = (const float*)d_in[3];
  const float* enc_Whh_f = (const float*)d_in[4];
  const float* enc_b_f   = (const float*)d_in[5];
  const float* enc_Wih_b = (const float*)d_in[6];
  const float* enc_Whh_b = (const float*)d_in[7];
  const float* enc_b_b   = (const float*)d_in[8];
  const float* dec_Wih   = (const float*)d_in[9];
  const float* dec_Whh   = (const float*)d_in[10];
  const float* dec_b     = (const float*)d_in[11];
  const float* attn_W    = (const float*)d_in[12];
  const float* mlp_W     = (const float*)d_in[13];
  const float* mlp_b     = (const float*)d_in[14];
  const float* out_W     = (const float*)d_in[15];
  const float* out_b     = (const float*)d_in[16];
  float* out = (float*)d_out;

  char* w = (char*)d_ws;
  size_t off = 0;
  auto alloc = [&](size_t nbytes) -> char* {
    off = (off + 255) & ~((size_t)255);
    char* p = w + off;
    off += nbytes;
    return p;
  };

  _Float16* embA16 = (_Float16*)alloc(8192ull * 512 * 2);
  _Float16* embN16 = (_Float16*)alloc(8208ull * 512 * 2);
  _Float16* wf16   = (_Float16*)alloc(1024ull * 512 * 2);
  _Float16* whf16  = (_Float16*)alloc(1024ull * 256 * 2);
  _Float16* wb16   = (_Float16*)alloc(1024ull * 512 * 2);
  _Float16* whb16  = (_Float16*)alloc(1024ull * 256 * 2);
  _Float16* wd16   = (_Float16*)alloc(2048ull * 512 * 2);
  _Float16* whd16  = (_Float16*)alloc(2048ull * 512 * 2);
  _Float16* aW16   = (_Float16*)alloc(512ull * 512 * 2);
  _Float16* mW16   = (_Float16*)alloc(512ull * 1024 * 2);
  _Float16* oW16   = (_Float16*)alloc((size_t)VPAD * 512 * 2);
  float*    obp    = (float*)alloc((size_t)VPAD * 4);
  float*    xWf    = (float*)alloc(8208ull * 1024 * 4);
  float*    xWb    = (float*)alloc(8208ull * 1024 * 4);
  float*    xWd    = (float*)alloc(8192ull * 2048 * 4);
  _Float16* enc16  = (_Float16*)alloc(8208ull * 512 * 2);
  _Float16* dec16  = (_Float16*)alloc(8192ull * 512 * 2);
  _Float16* full16 = (_Float16*)alloc(8192ull * 1024 * 2);
  float*    lossAcc = (float*)alloc(256);

  // post-encoder buffers aliased over dead xW projection buffers
  _Float16* proj16 = (_Float16*)xWf;                               // 8.4 MB
  float*    scores = (float*)((char*)xWf + 8208ull * 512 * 2);     // 16.8 MB
  _Float16* attnT  = (_Float16*)xWb;                               // 9.4 MB
  _Float16* encT   = (_Float16*)((char*)xWb + 16ull * 512 * SPAD * 2);
  _Float16* hid16  = (_Float16*)((char*)encT + 16ull * 512 * SPAD * 2);

  // --- init / weight conversion ---
  zero2_k<<<1, 32, 0, stream>>>(lossAcc);
  auto cvt = [&](const float* s, _Float16* d, size_t n) {
    f32_to_f16_k<<<(unsigned)((n + 255) / 256), 256, 0, stream>>>(s, d, n);
  };
  cvt(enc_Wih_f, wf16, 1024ull * 512);
  cvt(enc_Whh_f, whf16, 1024ull * 256);
  cvt(enc_Wih_b, wb16, 1024ull * 512);
  cvt(enc_Whh_b, whb16, 1024ull * 256);
  cvt(dec_Wih, wd16, 2048ull * 512);
  cvt(dec_Whh, whd16, 2048ull * 512);
  cvt(attn_W, aW16, 512ull * 512);
  cvt(mlp_W, mW16, 512ull * 1024);
  cvt(out_W, oW16, (size_t)VOCAB * 512);
  zero_f16_k<<<(unsigned)(((size_t)(VPAD - VOCAB) * 512 + 255) / 256), 256, 0,
               stream>>>(oW16 + (size_t)VOCAB * 512, (size_t)(VPAD - VOCAB) * 512);
  pad_bias_k<<<(VPAD + 255) / 256, 256, 0, stream>>>(out_b, obp, VOCAB, VPAD);

  // --- embedding gathers (f32 -> f16 rows) ---
  gather_emb_k<<<8192, 128, 0, stream>>>(emb, numeric, 512, embA16);
  gather_emb_k<<<8208, 128, 0, stream>>>(emb, noised, 513, embN16);

  // --- input projections: xW = x @ Wih^T + b  (f32 out) ---
  gemm_nt<<<dim3(513, 16, 1), 32, 0, stream>>>(embN16, 0, 512, wf16, 0, 512,
                                               xWf, 0, 1024, enc_b_f, 8208, 512, 0, 0);
  gemm_nt<<<dim3(513, 16, 1), 32, 0, stream>>>(embN16, 0, 512, wb16, 0, 512,
                                               xWb, 0, 1024, enc_b_b, 8208, 512, 0, 0);
  gemm_nt<<<dim3(512, 32, 1), 32, 0, stream>>>(embA16, 0, 512, wd16, 0, 512,
                                               xWd, 0, 2048, dec_b, 8192, 512, 0, 0);

  // --- LSTMs (persistent, serial over time) ---
  lstm_kernel<256><<<1, 256, 0, stream>>>(xWf, whf16, 513, 0, +1,
                                          enc16, 512, 0, ((_Float16*)0), 0, 0);
  lstm_kernel<256><<<1, 256, 0, stream>>>(xWb, whb16, 513, 512, -1,
                                          enc16, 512, 256, ((_Float16*)0), 0, 0);
  lstm_kernel<512><<<1, 512, 0, stream>>>(xWd, whd16, 512, 0, +1,
                                          dec16, 512, 0, full16, 1024, 0);

  // --- attention ---
  // proj[s,b,:] = enc[s,b,:] @ attn_W^T  (f16 out)
  gemm_nt<<<dim3(513, 8, 1), 32, 0, stream>>>(enc16, 0, 512, aW16, 0, 512,
                                              proj16, 0, 512, nullptr, 8208, 512, 0, 1);
  // scores[b,s,t] = sum_h proj[s,b,h]*dec[t,b,h]  (batched strided NT, f32 out)
  gemm_nt<<<dim3(33, 8, 16), 32, 0, stream>>>(proj16, 512, 8192, dec16, 512, 8192,
                                              scores, (long)513 * 512, 512, nullptr,
                                              513, 512, 0, 0);
  attn_softmax_k<<<32, 256, 0, stream>>>(scores, attnT);
  transpose_enc_k<<<(unsigned)((16ull * 512 * SPAD + 255) / 256), 256, 0,
                    stream>>>(enc16, encT);
  // ctx[t,b,h] = sum_s attnT[b,t,s]*encT[b,h,s] -> full16[...,512+h] (f16 out)
  gemm_nt<<<dim3(32, 8, 16), 32, 0, stream>>>(attnT, (long)512 * SPAD, SPAD,
                                              encT, (long)512 * SPAD, SPAD,
                                              full16 + 512, 1024, 16384, nullptr,
                                              512, SPAD, 0, 1);

  // --- MLP: hid = relu(full @ mlp_W^T + mlp_b)  (f16 out) ---
  gemm_nt<<<dim3(512, 8, 1), 32, 0, stream>>>(full16, 0, 1024, mW16, 0, 1024,
                                              hid16, 0, 512, mlp_b, 8192, 1024, 1, 1);

  // --- fused logits + log-softmax + masked NLL ---
  logits_nll<<<512, 256, 0, stream>>>(hid16, oW16, obp, numeric, VPAD, lossAcc);

  finalize_k<<<1, 1, 0, stream>>>(lossAcc, out);
}
